// SupervisedContrastiveLoss_77446850281771
// MI455X (gfx1250) — matrix operand order repaired and checked by reading, not compiled, not run on hardware
//
#include <hip/hip_runtime.h>

typedef __attribute__((ext_vector_type(16))) __bf16 v16bf;
typedef __attribute__((ext_vector_type(8)))  float  v8f;

#define NROWS 8192
#define DIM 256
#define BM 128
#define BN 64
#define LDSTRIDE 264               // padded row stride (bf16 elems) to spread LDS banks
#define NCOLSPLIT 8
#define COLS_PER_SPLIT (NROWS / NCOLSPLIT)   // 1024
#define INV_T 14.2857142857142857f

union BFrag  { uint4 q[2]; v16bf v; };
union BStore { __bf16 h[8]; uint4 q; };

// ---------------------------------------------------------------------------
// Kernel 1: L2-normalize each row and emit bf16 copy. One wave32 per row.
// ---------------------------------------------------------------------------
__global__ __launch_bounds__(256)
void scl_normalize_kernel(const float* __restrict__ feat,
                          unsigned short* __restrict__ fnb16) {
  __bf16* fnbf = reinterpret_cast<__bf16*>(fnb16);
  const int wave = threadIdx.x >> 5, lane = threadIdx.x & 31;
  const int row  = blockIdx.x * 8 + wave;
  const float4* src = reinterpret_cast<const float4*>(feat + (size_t)row * DIM + lane * 8);
  float4 u0 = src[0], u1 = src[1];
  float ss = u0.x*u0.x + u0.y*u0.y + u0.z*u0.z + u0.w*u0.w
           + u1.x*u1.x + u1.y*u1.y + u1.z*u1.z + u1.w*u1.w;
  #pragma unroll
  for (int m = 1; m < 32; m <<= 1) ss += __shfl_xor(ss, m, 32);
  const float inv = 1.0f / fmaxf(sqrtf(ss), 1e-6f);
  BStore o;
  o.h[0] = (__bf16)(u0.x*inv); o.h[1] = (__bf16)(u0.y*inv);
  o.h[2] = (__bf16)(u0.z*inv); o.h[3] = (__bf16)(u0.w*inv);
  o.h[4] = (__bf16)(u1.x*inv); o.h[5] = (__bf16)(u1.y*inv);
  o.h[6] = (__bf16)(u1.z*inv); o.h[7] = (__bf16)(u1.w*inv);
  *reinterpret_cast<uint4*>(fnbf + (size_t)row * DIM + lane * 8) = o.q;
}

// ---------------------------------------------------------------------------
// Kernel 2: fused GEMM (bf16 WMMA, fp32 acc) + masked fixed-max softmax stats.
// Grid: (NROWS/BM row blocks) x (NCOLSPLIT column splits), 256 threads.
// Each wave owns a 16-row strip; A strip kept in registers; B panels in LDS.
// ---------------------------------------------------------------------------
__global__ __launch_bounds__(256)
void scl_main_kernel(const unsigned short* __restrict__ fnb16,
                     const long long* __restrict__ labels,
                     float* __restrict__ srowP,
                     float* __restrict__ psumP,
                     float* __restrict__ pcntP) {
  const __bf16* fnbf = reinterpret_cast<const __bf16*>(fnb16);
  __shared__ __bf16 Bsh[BN * LDSTRIDE];

  const int lane = threadIdx.x & 31;
  const int wave = threadIdx.x >> 5;
  const int hi   = lane >> 4;         // lane-half
  const int hi8  = hi * 8;
  const int hi16 = hi * 16;
  const int nlo  = lane & 15;

  const int rowbase  = blockIdx.x * BM + wave * 16;
  const int colstart = blockIdx.y * COLS_PER_SPLIT;

  // --- A fragments: 16 rows x 256 K, resident in VGPRs (8 x v16bf).
  // ISA A-layout (16-bit 16x32): lane m holds K in [hi*8,hi*8+8) U [16+hi*8,...).
  BFrag a[8];
  {
    const __bf16* ap = fnbf + (size_t)(rowbase + nlo) * DIM;
    #pragma unroll
    for (int c = 0; c < 8; ++c) {
      a[c].q[0] = *reinterpret_cast<const uint4*>(ap + c*32 + hi8);
      a[c].q[1] = *reinterpret_cast<const uint4*>(ap + c*32 + 16 + hi8);
    }
  }

  // C/D layout: VGPR r -> row (r + 8*hi), col = lane&15.
  int irow[8]; int rlab[8];
  #pragma unroll
  for (int r = 0; r < 8; ++r) {
    irow[r] = rowbase + r + 8*hi;
    rlab[r] = (int)labels[irow[r]];
  }

  float sacc[8], ppos[8], pcnt[8];
  #pragma unroll
  for (int r = 0; r < 8; ++r) { sacc[r] = 0.f; ppos[r] = 0.f; pcnt[r] = 0.f; }

  const int jl  = threadIdx.x >> 2;   // staged row within panel (0..63)
  const int seg = threadIdx.x & 3;    // 64-element segment of that row

  for (int cb = colstart; cb < colstart + COLS_PER_SPLIT; cb += BN) {
    __syncthreads();
    { // cooperative stage of a 64-column panel (64 x 256 bf16 = 32 KB)
      const uint4* g = reinterpret_cast<const uint4*>(fnbf + (size_t)(cb + jl) * DIM + seg * 64);
      uint4* d = reinterpret_cast<uint4*>(&Bsh[jl * LDSTRIDE + seg * 64]);
      #pragma unroll
      for (int q = 0; q < 8; ++q) d[q] = g[q];
      if (cb + BN < colstart + COLS_PER_SPLIT)
        __builtin_prefetch(fnbf + (size_t)(cb + BN + jl) * DIM + seg * 64, 0, 1);
    }
    __syncthreads();

    // hoist this panel's per-lane column labels (4 loads in flight together)
    int jlab4[4];
    #pragma unroll
    for (int t = 0; t < 4; ++t) jlab4[t] = (int)labels[cb + t*16 + nlo];

    #pragma unroll
    for (int t = 0; t < 4; ++t) {           // four 16-column tiles per panel
      const int jt = t * 16;
      const __bf16* bp = &Bsh[(jt + nlo) * LDSTRIDE];

      // load ALL eight B fragments first (16x ds_load_b128, single dscnt wait),
      // then issue the 8 WMMAs back-to-back on the XDL pipe.
      BFrag b[8];   // B-layout (32x16): lane n holds col n, K in [hi*8,hi*8+8) U [16+hi*8,16+hi*8+8)
      #pragma unroll
      for (int c = 0; c < 8; ++c) {
        b[c].q[0] = *reinterpret_cast<const uint4*>(bp + c*32 + hi8);
        b[c].q[1] = *reinterpret_cast<const uint4*>(bp + c*32 + 16 + hi8);
      }
      v8f cacc = {0.f,0.f,0.f,0.f,0.f,0.f,0.f,0.f};
      #pragma unroll
      for (int c = 0; c < 8; ++c) {         // K = 256 in steps of 32
        cacc = __builtin_amdgcn_wmma_f32_16x16x32_bf16(false, a[c].v, false, b[c].v,
                                                       (short)0, cacc, false, false);
      }

      // branchless epilogue: scale, clip, self-mask, softmax + positive stats
      const int j = cb + jt + nlo;
      const int jlab = jlab4[t];
      #pragma unroll
      for (int r = 0; r < 8; ++r) {
        float x = cacc[r] * INV_T;
        x = fminf(fmaxf(x, -10.0f), 10.0f);
        const float e = __expf(x - 10.0f);        // clip bounds max => fixed max 10
        const bool nself = (j != irow[r]);        // self acts as -inf (excluded)
        const bool pos   = nself & (jlab == rlab[r]);
        sacc[r] += nself ? e   : 0.0f;
        ppos[r] += pos   ? x   : 0.0f;
        pcnt[r] += pos   ? 1.f : 0.0f;
      }
    }
  }

  // reduce the 16 lanes that share each row; write deterministic partials
  #pragma unroll
  for (int r = 0; r < 8; ++r) {
    float s = sacc[r], ps = ppos[r], pc = pcnt[r];
    #pragma unroll
    for (int m = 1; m < 16; m <<= 1) {
      s  += __shfl_xor(s,  m, 32);
      ps += __shfl_xor(ps, m, 32);
      pc += __shfl_xor(pc, m, 32);
    }
    if (nlo == 0) {
      const size_t idx = (size_t)blockIdx.y * NROWS + irow[r];
      srowP[idx] = s; psumP[idx] = ps; pcntP[idx] = pc;
    }
  }
}

// ---------------------------------------------------------------------------
// Kernel 3: combine column-split partials (fixed order => deterministic),
// per-row loss, reduce to scalar.
// ---------------------------------------------------------------------------
__global__ __launch_bounds__(256)
void scl_finalize_kernel(const float* __restrict__ srowP,
                         const float* __restrict__ psumP,
                         const float* __restrict__ pcntP,
                         float* __restrict__ out) {
  __shared__ float shs[8], shv[8];
  const int tid = threadIdx.x;
  float ls = 0.f, lv = 0.f;
  for (int i = tid; i < NROWS; i += 256) {
    float s = 0.f, ps = 0.f, pc = 0.f;
    #pragma unroll
    for (int cs = 0; cs < NCOLSPLIT; ++cs) {
      s  += srowP[(size_t)cs * NROWS + i];
      ps += psumP[(size_t)cs * NROWS + i];
      pc += pcntP[(size_t)cs * NROWS + i];
    }
    if (pc > 0.5f) {
      const float L = 10.0f + __logf(s);            // logsumexp (fixed max 10)
      float mlp = ps / pc - L;                      // mean log prob of positives
      mlp = fminf(fmaxf(mlp, -50.0f), 50.0f);       // (no-op clip, kept for fidelity)
      ls += -mlp;                                   // T/BASE_T == 1
      lv += 1.0f;
    }
  }
  #pragma unroll
  for (int m = 1; m < 32; m <<= 1) { ls += __shfl_xor(ls, m, 32); lv += __shfl_xor(lv, m, 32); }
  if ((tid & 31) == 0) { shs[tid >> 5] = ls; shv[tid >> 5] = lv; }
  __syncthreads();
  if (tid == 0) {
    float S = 0.f, V = 0.f;
    #pragma unroll
    for (int k = 0; k < 8; ++k) { S += shs[k]; V += shv[k]; }
    out[0] = (V > 0.f) ? (S / fmaxf(V, 1.0f)) : 0.f;
  }
}

// ---------------------------------------------------------------------------
extern "C" void kernel_launch(void* const* d_in, const int* in_sizes, int n_in,
                              void* d_out, int out_size, void* d_ws, size_t ws_size,
                              hipStream_t stream) {
  (void)in_sizes; (void)n_in; (void)out_size; (void)ws_size;
  const float*     features = (const float*)d_in[0];
  const long long* labels   = (const long long*)d_in[1];   // jnp int64
  float* out = (float*)d_out;

  char* ws = (char*)d_ws;
  unsigned short* fnbf = (unsigned short*)ws;                          // 4 MiB bf16 normalized copy
  float* srowP = (float*)(ws + (size_t)NROWS * DIM * 2);               // 8 x 8192 f32
  float* psumP = srowP + (size_t)NCOLSPLIT * NROWS;                    // 8 x 8192 f32
  float* pcntP = psumP + (size_t)NCOLSPLIT * NROWS;                    // 8 x 8192 f32

  scl_normalize_kernel<<<NROWS / 8, 256, 0, stream>>>(features, fnbf);
  scl_main_kernel<<<dim3(NROWS / BM, NCOLSPLIT), 256, 0, stream>>>(fnbf, labels,
                                                                   srowP, psumP, pcntP);
  scl_finalize_kernel<<<1, 256, 0, stream>>>(srowP, psumP, pcntP, out);
}
